// GIN_KAN_69097433858366
// MI455X (gfx1250) — matrix operand order repaired and checked by reading, not compiled, not run on hardware
//
#include <hip/hip_runtime.h>
#include <hip/hip_bf16.h>
#include <math.h>

// ---------------------------------------------------------------------------
// GIN-KAN forward for MI455X (gfx1250, wave32).
// Heavy GEMMs via V_WMMA_F32_16X16X32_BF16, register-blocked 32x32 C per wave
// (4 WMMAs per 4 fragment loads -> 2 VMEM issues per matrix op).
// ---------------------------------------------------------------------------

#define N_NODES   20000
#define N_EDGES   50000
#define HIDD      128
#define OUT_DIM   10
#define EDGE_CHUNK 10000   // 625 * 16, exact 16-row tiles

typedef __attribute__((ext_vector_type(16))) __bf16 v16bf;
typedef __attribute__((ext_vector_type(8)))  __bf16 v8bf;
typedef __attribute__((ext_vector_type(8)))  float  v8f;

// -------------------- cubic B-spline basis (8 coefs, uniform knots) --------
__device__ __forceinline__ void bspline8(float x, float* __restrict__ out) {
  const float h = 0.4f;
  float t[12];
#pragma unroll
  for (int j = 0; j < 12; ++j) t[j] = (float)(j - 3) * h - 1.0f;
  float b[11];
#pragma unroll
  for (int j = 0; j < 11; ++j) b[j] = (x >= t[j] && x < t[j + 1]) ? 1.0f : 0.0f;
#pragma unroll
  for (int k = 1; k <= 3; ++k) {
#pragma unroll
    for (int j = 0; j + k < 11; ++j) {
      float left  = (x - t[j]) / (t[j + k] - t[j]);
      float right = (t[j + k + 1] - x) / (t[j + k + 1] - t[j + 1]);
      b[j] = left * b[j] + right * b[j + 1];
    }
  }
#pragma unroll
  for (int j = 0; j < 8; ++j) out[j] = b[j];
}

// -------------------- WMMA GEMM: C[M,N] = A[M,K] * B[N,K]^T + bias ---------
// A,B bf16 row-major; M%16==0, N%32==0, K%32==0.
// One wave owns a 32x32 C block (2x2 tiles); wave-uniform tail handles M%32.
__global__ void wmma_gemm_bf16(const __bf16* __restrict__ A,
                               const __bf16* __restrict__ B,
                               const float*  __restrict__ bias,  // [N] or null
                               float* __restrict__ C,
                               int M, int N, int K)
{
  const int lane = threadIdx.x & 31;
  const int wave = threadIdx.x >> 5;
  const int wpb  = blockDim.x >> 5;
  const int mp   = blockIdx.y * wpb + wave;   // 32-row block index
  const int mts  = M >> 4;                    // # of 16-row tiles
  const int mt0  = mp * 2;
  if (mt0 >= mts) return;                     // wave-uniform: EXEC all-ones
  const bool two_m = (mt0 + 1) < mts;
  const int nt0  = blockIdx.x * 2;
  const int half = lane >> 4;
  const int l16  = lane & 15;

  const size_t arow0 = (size_t)(mt0 * 16 + l16) * (size_t)K;
  const size_t arow1 = arow0 + (size_t)16 * (size_t)K;
  const size_t brow0 = (size_t)(nt0 * 16 + l16) * (size_t)K;
  const size_t brow1 = brow0 + (size_t)16 * (size_t)K;

  // A 16x32 fragment (ISA layout): lane row m=l16,
  // K = {half*8..half*8+7} then {16+half*8..16+half*8+7}
  auto loadA = [&](size_t arow, int k0) -> v16bf {
    v8bf lo = *(const v8bf*)(A + arow + k0 + half * 8);
    v8bf hi = *(const v8bf*)(A + arow + k0 + 16 + half * 8);
    return __builtin_shufflevector(lo, hi,
             0,1,2,3,4,5,6,7,8,9,10,11,12,13,14,15);
  };
  // B 32x16 fragment: lane col n=l16,
  // K = {half*8..half*8+7} then {16+half*8..16+half*8+7}
  auto loadB = [&](size_t brow, int k0) -> v16bf {
    v8bf lo = *(const v8bf*)(B + brow + k0 + half * 8);
    v8bf hi = *(const v8bf*)(B + brow + k0 + 16 + half * 8);
    return __builtin_shufflevector(lo, hi,
             0,1,2,3,4,5,6,7,8,9,10,11,12,13,14,15);
  };
  auto storeT = [&](v8f acc, int mt, int nt) {
    const int n  = nt * 16 + l16;
    const float bv = bias ? bias[n] : 0.0f;
    const int m0 = mt * 16 + half * 8;        // C: VGPR v -> row m0+v, col n
#pragma unroll
    for (int v = 0; v < 8; ++v)
      C[(size_t)(m0 + v) * (size_t)N + n] = acc[v] + bv;
  };

  v8f acc00 = {0.f,0.f,0.f,0.f,0.f,0.f,0.f,0.f};
  v8f acc01 = acc00, acc10 = acc00, acc11 = acc00;

  if (two_m) {
    for (int k0 = 0; k0 < K; k0 += 32) {
      v16bf a0 = loadA(arow0, k0);
      v16bf a1 = loadA(arow1, k0);
      v16bf b0 = loadB(brow0, k0);
      v16bf b1 = loadB(brow1, k0);
      acc00 = __builtin_amdgcn_wmma_f32_16x16x32_bf16(
                  false, a0, false, b0, (short)0, acc00, false, false);
      acc01 = __builtin_amdgcn_wmma_f32_16x16x32_bf16(
                  false, a0, false, b1, (short)0, acc01, false, false);
      acc10 = __builtin_amdgcn_wmma_f32_16x16x32_bf16(
                  false, a1, false, b0, (short)0, acc10, false, false);
      acc11 = __builtin_amdgcn_wmma_f32_16x16x32_bf16(
                  false, a1, false, b1, (short)0, acc11, false, false);
    }
    storeT(acc00, mt0, nt0);
    storeT(acc01, mt0, nt0 + 1);
    storeT(acc10, mt0 + 1, nt0);
    storeT(acc11, mt0 + 1, nt0 + 1);
  } else {
    for (int k0 = 0; k0 < K; k0 += 32) {
      v16bf a0 = loadA(arow0, k0);
      v16bf b0 = loadB(brow0, k0);
      v16bf b1 = loadB(brow1, k0);
      acc00 = __builtin_amdgcn_wmma_f32_16x16x32_bf16(
                  false, a0, false, b0, (short)0, acc00, false, false);
      acc01 = __builtin_amdgcn_wmma_f32_16x16x32_bf16(
                  false, a0, false, b1, (short)0, acc01, false, false);
    }
    storeT(acc00, mt0, nt0);
    storeT(acc01, mt0, nt0 + 1);
  }
}

// -------------------- elementwise / scatter kernels ------------------------
__global__ void init_pooled_kernel(const float* __restrict__ h,
                                   const float* __restrict__ eps, int layer,
                                   float* __restrict__ pooled, int total)
{
  int i = blockIdx.x * blockDim.x + threadIdx.x;
  if (i < total) pooled[i] = (1.0f + eps[layer]) * h[i];
}

__global__ void edge_aggregate_kernel(const float* __restrict__ h,
                                      const int* __restrict__ src,
                                      const int* __restrict__ dst,
                                      float* __restrict__ pooled, int E)
{
  int e = blockIdx.x;
  int f = threadIdx.x;
  if (e < E)
    atomicAdd(&pooled[(size_t)src[e] * HIDD + f],
              h[(size_t)dst[e] * HIDD + f]);
}

__global__ void edge_scatter_alpha_kernel(const float* __restrict__ hkan,
                                          const int* __restrict__ src,
                                          const int* __restrict__ dst,
                                          const float* __restrict__ alpha,
                                          float* __restrict__ h_att, int E)
{
  int e = blockIdx.x;
  int f = threadIdx.x;
  if (e < E)
    atomicAdd(&h_att[(size_t)src[e] * HIDD + f],
              alpha[e] * hkan[(size_t)dst[e] * HIDD + f]);
}

// expand X[r+row0, 0:in_dim] -> A[r, 0:in_dim*9] = [silu(x) | bases(x)] (bf16)
__global__ void kan_expand_kernel(const float* __restrict__ X,
                                  __bf16* __restrict__ A,
                                  int rows, int in_dim, int row0)
{
  int idx = blockIdx.x * blockDim.x + threadIdx.x;
  int total = rows * in_dim;
  if (idx >= total) return;
  int r = idx / in_dim, i = idx - r * in_dim;
  float x = X[(size_t)(r + row0) * in_dim + i];
  float s = x / (1.0f + expf(-x));            // silu
  size_t arow = (size_t)r * (size_t)(in_dim * 9);
  A[arow + i] = (__bf16)s;
  float bs[8]; bspline8(x, bs);
  __bf16* p = A + arow + in_dim + (size_t)i * 8;
#pragma unroll
  for (int k = 0; k < 8; ++k) p[k] = (__bf16)bs[k];
}

// pack [base_w | spline_w] fp32 -> bf16 row-major [out, in*9] matching expand
__global__ void pack_kan_w_kernel(const float* __restrict__ base_w,   // [o,i]
                                  const float* __restrict__ spline_w, // [o,i,8]
                                  __bf16* __restrict__ W, int out_dim, int in_dim)
{
  int idx = blockIdx.x * blockDim.x + threadIdx.x;
  int total = out_dim * in_dim;
  if (idx >= total) return;
  int o = idx / in_dim, i = idx - o * in_dim;
  size_t wrow = (size_t)o * (size_t)(in_dim * 9);
  W[wrow + i] = (__bf16)base_w[(size_t)o * in_dim + i];
  const float* sp = spline_w + ((size_t)o * in_dim + i) * 8;
  __bf16* d = W + wrow + in_dim + (size_t)i * 8;
#pragma unroll
  for (int k = 0; k < 8; ++k) d[k] = (__bf16)sp[k];
}

__global__ void f32_to_bf16_kernel(const float* __restrict__ s,
                                   __bf16* __restrict__ d, int n)
{
  int i = blockIdx.x * blockDim.x + threadIdx.x;
  if (i < n) d[i] = (__bf16)s[i];
}

__global__ void gather_rows_bf16_kernel(const __bf16* __restrict__ mat,
                                        const int* __restrict__ idxv,
                                        __bf16* __restrict__ out, int E)
{
  int idx = blockIdx.x * blockDim.x + threadIdx.x;
  if (idx >= E * HIDD) return;
  int e = idx / HIDD, f = idx - e * HIDD;
  out[idx] = mat[(size_t)idxv[e] * HIDD + f];
}

// -------------------- BatchNorm (training-mode batch stats) ----------------
__global__ void bn_stats_kernel(const float* __restrict__ X,
                                float* __restrict__ mean,
                                float* __restrict__ var, int M)
{
  __shared__ float ssum[256];
  __shared__ float ssq[256];
  int f = blockIdx.x;                        // feature 0..127
  float s = 0.f, q = 0.f;
  for (int r = threadIdx.x; r < M; r += blockDim.x) {
    float v = X[(size_t)r * HIDD + f];
    s += v; q += v * v;
  }
  ssum[threadIdx.x] = s; ssq[threadIdx.x] = q;
  __syncthreads();
  for (int st = 128; st > 0; st >>= 1) {
    if (threadIdx.x < st) {
      ssum[threadIdx.x] += ssum[threadIdx.x + st];
      ssq[threadIdx.x]  += ssq[threadIdx.x + st];
    }
    __syncthreads();
  }
  if (threadIdx.x == 0) {
    float m = ssum[0] / (float)M;
    mean[f] = m;
    var[f]  = ssq[0] / (float)M - m * m;
  }
}

__global__ void bn_relu_kernel(float* __restrict__ X,
                               const float* __restrict__ mean,
                               const float* __restrict__ var,
                               const float* __restrict__ gamma,
                               const float* __restrict__ beta,
                               __bf16* __restrict__ Xbf, int total)
{
  int i = blockIdx.x * blockDim.x + threadIdx.x;
  if (i >= total) return;
  int f = i & (HIDD - 1);
  float v = (X[i] - mean[f]) * rsqrtf(var[f] + 1e-5f) * gamma[f] + beta[f];
  v = fmaxf(v, 0.0f);
  X[i] = v;
  Xbf[i] = (__bf16)v;
}

// -------------------- attention epilogue -----------------------------------
// Per edge: logits over 4 heads from (q+pos).(k+pos), softmax over heads,
// Z = [h_kan[src] | w*v]  (fp32, [E,256])
__global__ void attn_edge_kernel(const float* __restrict__ Q,
                                 const float* __restrict__ Km,
                                 const float* __restrict__ V,
                                 const float* __restrict__ pos,   // [128]
                                 const float* __restrict__ hkan,
                                 const int* __restrict__ src,
                                 float* __restrict__ Z, int E)
{
  int e = blockIdx.x * blockDim.x + threadIdx.x;
  if (e >= E) return;
  const float scale = 0.17677669529663687f;  // 1/sqrt(32)
  float logit[4];
#pragma unroll
  for (int hh = 0; hh < 4; ++hh) {
    float acc = 0.f;
    for (int d = 0; d < 32; ++d) {
      float p = pos[hh * 32 + d];
      float qv = Q [(size_t)e * HIDD + hh * 32 + d] + p;
      float kv = Km[(size_t)e * HIDD + hh * 32 + d] + p;
      acc += qv * kv;
    }
    logit[hh] = acc * scale;
  }
  float mx = fmaxf(fmaxf(logit[0], logit[1]), fmaxf(logit[2], logit[3]));
  float w[4], s = 0.f;
#pragma unroll
  for (int hh = 0; hh < 4; ++hh) { w[hh] = expf(logit[hh] - mx); s += w[hh]; }
  float inv = 1.0f / s;
  int sn = src[e];
  float* z = Z + (size_t)e * 256;
  for (int i = 0; i < HIDD; ++i) z[i] = hkan[(size_t)sn * HIDD + i];
#pragma unroll
  for (int hh = 0; hh < 4; ++hh) {
    float ww = w[hh] * inv;
    for (int d = 0; d < 32; ++d)
      z[HIDD + hh * 32 + d] = ww * V[(size_t)e * HIDD + hh * 32 + d];
  }
}

// gelu(z1) -> KAN(128->1) -> sigmoid : edge gate alpha
__global__ void gelu_kan2_sigmoid_kernel(const float* __restrict__ z1, // [E,128]
                                         const float* __restrict__ bw, // [128]
                                         const float* __restrict__ sw, // [128*8]
                                         float* __restrict__ alpha, int E)
{
  int e = blockIdx.x * blockDim.x + threadIdx.x;
  if (e >= E) return;
  float acc = 0.f;
  for (int i = 0; i < HIDD; ++i) {
    float x = z1[(size_t)e * HIDD + i];
    float g = 0.5f * x * (1.0f + erff(x * 0.7071067811865476f)); // exact gelu
    float sl = g / (1.0f + expf(-g));                            // silu
    acc += sl * bw[i];
    float bs[8]; bspline8(g, bs);
#pragma unroll
    for (int k = 0; k < 8; ++k) acc += bs[k] * sw[i * 8 + k];
  }
  alpha[e] = 1.0f / (1.0f + expf(-acc));
}

// -------------------- add residual + LayerNorm -----------------------------
__global__ void add_layernorm_kernel(const float* __restrict__ hatt,
                                     const float* __restrict__ resid,
                                     const float* __restrict__ gamma,
                                     const float* __restrict__ beta,
                                     float* __restrict__ out)
{
  __shared__ float red[HIDD];
  int n = blockIdx.x, f = threadIdx.x;
  size_t base = (size_t)n * HIDD;
  float v = hatt[base + f] + resid[base + f];
  red[f] = v;
  __syncthreads();
  for (int st = 64; st > 0; st >>= 1) {
    if (f < st) red[f] += red[f + st];
    __syncthreads();
  }
  float mu = red[0] * (1.0f / HIDD);
  __syncthreads();
  float d = v - mu;
  red[f] = d * d;
  __syncthreads();
  for (int st = 64; st > 0; st >>= 1) {
    if (f < st) red[f] += red[f + st];
    __syncthreads();
  }
  float var = red[0] * (1.0f / HIDD);
  out[base + f] = d * rsqrtf(var + 1e-5f) * gamma[f] + beta[f];
}

// -------------------- classifier -------------------------------------------
__global__ void classifier_kernel(const float* __restrict__ h,
                                  const float* __restrict__ W,  // [10,128]
                                  const float* __restrict__ b,  // [10]
                                  float* __restrict__ out, int Mn)
{
  int idx = blockIdx.x * blockDim.x + threadIdx.x;
  if (idx >= Mn * OUT_DIM) return;
  int n = idx / OUT_DIM, c = idx - n * OUT_DIM;
  float a = b[c];
  for (int i = 0; i < HIDD; ++i)
    a += h[(size_t)n * HIDD + i] * W[(size_t)c * HIDD + i];
  out[idx] = a;
}

// ===========================================================================
extern "C" void kernel_launch(void* const* d_in, const int* in_sizes, int n_in,
                              void* d_out, int out_size, void* d_ws, size_t ws_size,
                              hipStream_t stream)
{
  (void)in_sizes; (void)n_in; (void)out_size; (void)ws_size;
  const float* x        = (const float*)d_in[0];
  const int*   src      = (const int*)d_in[1];
  const int*   dst      = (const int*)d_in[2];
  const float* eps      = (const float*)d_in[3];
  const float* kan_base = (const float*)d_in[4];   // [2,128,128]
  const float* kan_spl  = (const float*)d_in[5];   // [2,128,128,8]
  const float* bn_gamma = (const float*)d_in[6];
  const float* bn_beta  = (const float*)d_in[7];
  const float* ln_gamma = (const float*)d_in[8];
  const float* ln_beta  = (const float*)d_in[9];
  const float* qW = (const float*)d_in[10];
  const float* qb = (const float*)d_in[11];
  const float* kW = (const float*)d_in[12];
  const float* kb = (const float*)d_in[13];
  const float* vW = (const float*)d_in[14];
  const float* vb = (const float*)d_in[15];
  const float* pos = (const float*)d_in[16];       // [2,4,32]
  const float* ak1b = (const float*)d_in[17];      // [2,128,256]
  const float* ak1s = (const float*)d_in[18];      // [2,128,256,8]
  const float* ak2b = (const float*)d_in[19];      // [2,1,128]
  const float* ak2s = (const float*)d_in[20];      // [2,1,128,8]
  const float* skip_W = (const float*)d_in[21];
  const float* skip_b = (const float*)d_in[22];
  const float* clf_W  = (const float*)d_in[23];
  const float* clf_b  = (const float*)d_in[24];

  // ---- workspace carve (bump allocator, 256B aligned) ----
  char* ws = (char*)d_ws;
  size_t off = 0;
  auto alloc = [&](size_t bytes) -> void* {
    off = (off + 255) & ~(size_t)255;
    void* p = ws + off;
    off += bytes;
    return p;
  };
  const size_t NH  = (size_t)N_NODES * HIDD;
  const size_t EH  = (size_t)N_EDGES * HIDD;
  float*  pooled  = (float*) alloc(NH * 4);
  __bf16* Aexp    = (__bf16*)alloc((size_t)N_NODES * 1152 * 2); // == EDGE_CHUNK*2304*2
  __bf16* Wkan    = (__bf16*)alloc((size_t)HIDD * 1152 * 2);
  float*  h_kan   = (float*) alloc(NH * 4);
  __bf16* hkan_bf = (__bf16*)alloc(NH * 2);
  float*  bn_mean = (float*) alloc(HIDD * 4);
  float*  bn_var  = (float*) alloc(HIDD * 4);
  __bf16* Wq      = (__bf16*)alloc((size_t)HIDD * HIDD * 2);
  __bf16* Wk      = (__bf16*)alloc((size_t)HIDD * HIDD * 2);
  __bf16* Wv      = (__bf16*)alloc((size_t)HIDD * HIDD * 2);
  __bf16* Wskip   = (__bf16*)alloc((size_t)HIDD * HIDD * 2);
  __bf16* Hsrc    = (__bf16*)alloc(EH * 2);
  __bf16* Hdst    = (__bf16*)alloc(EH * 2);
  float*  Qm      = (float*) alloc(EH * 4);
  float*  Km      = (float*) alloc(EH * 4);
  float*  Vm      = (float*) alloc(EH * 4);
  float*  Z       = (float*) alloc((size_t)N_EDGES * 256 * 4);
  __bf16* Wak1    = (__bf16*)alloc((size_t)HIDD * 2304 * 2);
  float*  z1      = (float*) alloc(EH * 4);
  float*  alphab  = (float*) alloc((size_t)N_EDGES * 4);
  float*  h_att   = (float*) alloc(NH * 4);
  float*  resid   = (float*) alloc(NH * 4);
  __bf16* Xbf     = (__bf16*)alloc(NH * 2);
  float*  hbuf    = (float*) alloc(NH * 4);

  const int TPB = 256;
  auto cdiv = [](int a, int b) { return (a + b - 1) / b; };
  // WMMA launch: grid.x = N/32, grid.y = ceil(ceil(M/32)/8), 8 waves/block
  auto wgrid = [&](int M, int N) { return dim3(N / 32, cdiv(cdiv(M, 32), 8)); };

  const float* h = x;       // current hidden state
  const float* h_prev = x;

  for (int l = 0; l < 2; ++l) {
    // ---- GIN aggregation: pooled = (1+eps)*h + segment_sum(h[dst] @ src) ----
    init_pooled_kernel<<<cdiv((int)NH, TPB), TPB, 0, stream>>>(
        h, eps, l, pooled, (int)NH);
    edge_aggregate_kernel<<<N_EDGES, HIDD, 0, stream>>>(
        h, src, dst, pooled, N_EDGES);

    // ---- node KAN: expand + one WMMA GEMM (K = 128 + 128*8 = 1152) ----
    pack_kan_w_kernel<<<cdiv(HIDD * HIDD, TPB), TPB, 0, stream>>>(
        kan_base + (size_t)l * HIDD * HIDD,
        kan_spl  + (size_t)l * HIDD * HIDD * 8, Wkan, HIDD, HIDD);
    kan_expand_kernel<<<cdiv(N_NODES * HIDD, TPB), TPB, 0, stream>>>(
        pooled, Aexp, N_NODES, HIDD, 0);
    wmma_gemm_bf16<<<wgrid(N_NODES, HIDD), TPB, 0, stream>>>(
        Aexp, Wkan, nullptr, h_kan, N_NODES, HIDD, 1152);

    // ---- BatchNorm (batch stats) + ReLU, keep bf16 copy for q/k/v ----
    bn_stats_kernel<<<HIDD, TPB, 0, stream>>>(h_kan, bn_mean, bn_var, N_NODES);
    bn_relu_kernel<<<cdiv((int)NH, TPB), TPB, 0, stream>>>(
        h_kan, bn_mean, bn_var, bn_gamma + l * HIDD, bn_beta + l * HIDD,
        hkan_bf, (int)NH);

    // ---- edge attention: gather, q/k/v WMMA GEMMs, softmax over heads ----
    f32_to_bf16_kernel<<<cdiv(HIDD * HIDD, TPB), TPB, 0, stream>>>(
        qW + (size_t)l * HIDD * HIDD, Wq, HIDD * HIDD);
    f32_to_bf16_kernel<<<cdiv(HIDD * HIDD, TPB), TPB, 0, stream>>>(
        kW + (size_t)l * HIDD * HIDD, Wk, HIDD * HIDD);
    f32_to_bf16_kernel<<<cdiv(HIDD * HIDD, TPB), TPB, 0, stream>>>(
        vW + (size_t)l * HIDD * HIDD, Wv, HIDD * HIDD);
    gather_rows_bf16_kernel<<<cdiv((int)EH, TPB), TPB, 0, stream>>>(
        hkan_bf, src, Hsrc, N_EDGES);
    gather_rows_bf16_kernel<<<cdiv((int)EH, TPB), TPB, 0, stream>>>(
        hkan_bf, dst, Hdst, N_EDGES);
    wmma_gemm_bf16<<<wgrid(N_EDGES, HIDD), TPB, 0, stream>>>(
        Hsrc, Wq, qb + l * HIDD, Qm, N_EDGES, HIDD, HIDD);
    wmma_gemm_bf16<<<wgrid(N_EDGES, HIDD), TPB, 0, stream>>>(
        Hdst, Wk, kb + l * HIDD, Km, N_EDGES, HIDD, HIDD);
    wmma_gemm_bf16<<<wgrid(N_EDGES, HIDD), TPB, 0, stream>>>(
        Hdst, Wv, vb + l * HIDD, Vm, N_EDGES, HIDD, HIDD);
    attn_edge_kernel<<<cdiv(N_EDGES, TPB), TPB, 0, stream>>>(
        Qm, Km, Vm, pos + l * HIDD, h_kan, src, Z, N_EDGES);

    // ---- edge KAN1 (K = 256 + 256*8 = 2304), chunked to bound expansion ----
    pack_kan_w_kernel<<<cdiv(HIDD * 256, TPB), TPB, 0, stream>>>(
        ak1b + (size_t)l * HIDD * 256,
        ak1s + (size_t)l * HIDD * 256 * 8, Wak1, HIDD, 256);
    for (int c0 = 0; c0 < N_EDGES; c0 += EDGE_CHUNK) {
      kan_expand_kernel<<<cdiv(EDGE_CHUNK * 256, TPB), TPB, 0, stream>>>(
          Z, Aexp, EDGE_CHUNK, 256, c0);
      wmma_gemm_bf16<<<wgrid(EDGE_CHUNK, HIDD), TPB, 0, stream>>>(
          Aexp, Wak1, nullptr, z1 + (size_t)c0 * HIDD,
          EDGE_CHUNK, HIDD, 2304);
    }

    // ---- gelu -> KAN2(128->1) -> sigmoid : edge gate ----
    gelu_kan2_sigmoid_kernel<<<cdiv(N_EDGES, TPB), TPB, 0, stream>>>(
        z1, ak2b + (size_t)l * HIDD, ak2s + (size_t)l * HIDD * 8,
        alphab, N_EDGES);

    // ---- h_att = h_kan + segment_sum(alpha * h_kan[dst]) ----
    hipMemcpyAsync(h_att, h_kan, NH * 4, hipMemcpyDeviceToDevice, stream);
    edge_scatter_alpha_kernel<<<N_EDGES, HIDD, 0, stream>>>(
        h_kan, src, dst, alphab, h_att, N_EDGES);

    // ---- residual + LayerNorm ----
    const float* residp;
    if (l == 0) {
      f32_to_bf16_kernel<<<cdiv((int)NH, TPB), TPB, 0, stream>>>(
          x, Xbf, (int)NH);
      f32_to_bf16_kernel<<<cdiv(HIDD * HIDD, TPB), TPB, 0, stream>>>(
          skip_W, Wskip, HIDD * HIDD);
      wmma_gemm_bf16<<<wgrid(N_NODES, HIDD), TPB, 0, stream>>>(
          Xbf, Wskip, skip_b, resid, N_NODES, HIDD, HIDD);
      residp = resid;
    } else {
      residp = h_prev;   // == hbuf (layer-0 output), untouched this layer
    }
    add_layernorm_kernel<<<N_NODES, HIDD, 0, stream>>>(
        h_att, residp, ln_gamma + l * HIDD, ln_beta + l * HIDD, hbuf);
    h = hbuf;
    h_prev = hbuf;
  }

  // ---- classifier ----
  classifier_kernel<<<cdiv(N_NODES * OUT_DIM, TPB), TPB, 0, stream>>>(
      hbuf, clf_W, clf_b, (float*)d_out, N_NODES);
}
